// EquivariantEmbeddingBlock_13898514170596
// MI455X (gfx1250) — matrix operand order repaired and checked by reading, not compiled, not run on hardware
//
#include <hip/hip_runtime.h>

typedef __attribute__((ext_vector_type(16))) __bf16 v16bf;
typedef __attribute__((ext_vector_type(8)))  float  v8f;
typedef __attribute__((ext_vector_type(4)))  float  v4f;

#define N_EDGES 300000
#define N_TILES (N_EDGES / 16)   // 18750
#define N_PAIRS (N_TILES / 2)    // 9375 (exact)
#define EMB     128
#define NZ      95

// Workspace layout (bytes, 128B-aligned offsets)
#define WS_EWI    0          // 95*128 f32  = 48640 B
#define WS_EWJ    48640      // 95*128 f32
#define WS_BFUSED 97280      // 128 f32 = 512 B
#define WS_VWI    97792      // 95*3*128 f32 = 145920 B
#define WS_VWJ    243712     // 95*3*128 f32
#define WS_BV2S   389632     // 96 frags * 512 bf16 = 98304 B
#define WS_BRBF   487936     // 8 frags * 512 bf16 = 8192 B   (contiguous after BV2S)
#define WS_TOTAL  496128

#define N_FRAGS   104                    // 96 v2s + 8 rbf
#define LDS_B_BYTES (N_FRAGS * 512 * 2)  // 106496

__device__ __forceinline__ __bf16 f2bf(float f) { return (__bf16)f; }

// ---------------- setup kernels (tiny, run once per launch) ----------------

__global__ void setup_ew(const float* __restrict__ emb, const float* __restrict__ Ws,
                         const float* __restrict__ brbf, const float* __restrict__ bs,
                         float* __restrict__ EWi, float* __restrict__ EWj,
                         float* __restrict__ bfused) {
  int idx = blockIdx.x * blockDim.x + threadIdx.x;
  const int NE = NZ * EMB;
  if (idx < NE) {
    int z = idx / EMB, n = idx % EMB;
    float s = 0.f;
    for (int k = 0; k < EMB; ++k) s += emb[z * EMB + k] * Ws[k * EMB + n];
    EWi[idx] = s;
  } else if (idx < 2 * NE) {
    int t = idx - NE;
    int z = t / EMB, n = t % EMB;
    float s = 0.f;
    for (int k = 0; k < EMB; ++k) s += emb[z * EMB + k] * Ws[(EMB + k) * EMB + n];
    EWj[t] = s;
  } else if (idx < 2 * NE + EMB) {
    int n = idx - 2 * NE;
    float s = bs[n];
    for (int k = 0; k < EMB; ++k) s += brbf[k] * Ws[(2 * EMB + k) * EMB + n];
    bfused[n] = s;
  }
}

__global__ void setup_vw(const float* __restrict__ vemb, const float* __restrict__ Wv,
                         float* __restrict__ VWi, float* __restrict__ VWj) {
  int idx = blockIdx.x * blockDim.x + threadIdx.x;
  const int NV = NZ * 3 * EMB;
  if (idx >= 2 * NV) return;
  int which = (idx >= NV) ? 1 : 0;
  int t = idx - which * NV;
  int z = t / (3 * EMB);
  int rem = t % (3 * EMB);
  int d = rem / EMB, c = rem % EMB;
  float s = 0.f;
  for (int k = 0; k < EMB; ++k)
    s += vemb[(z * EMB + k) * 3 + d] * Wv[(k + which * EMB) * EMB + c];
  (which ? VWj : VWi)[t] = s;
}

// Pre-swizzle B matrices into WMMA 16x16x32 bf16 fragment order.
// Fragment: 32 lanes x 16 bf16; lane half h = lane>>4; element t -> K_local = 16*(t>>3) + 8h + (t&7);
// column n = nt*16 + (lane&15).
__global__ void setup_bpack(const float* __restrict__ Wv2s, const float* __restrict__ Wrbf,
                            const float* __restrict__ Ws,
                            __bf16* __restrict__ Bv2s, __bf16* __restrict__ Brbf) {
  int idx = blockIdx.x * blockDim.x + threadIdx.x;
  if (idx < 96 * 512) {
    int frag = idx >> 9, within = idx & 511;
    int lane = within >> 4, t = within & 15;
    int nt = frag & 7, ktd = frag >> 3;
    int kt = ktd & 3, d = ktd >> 2;
    int c = 32 * kt + 16 * (t >> 3) + 8 * (lane >> 4) + (t & 7);
    int n = nt * 16 + (lane & 15);
    Bv2s[idx] = f2bf(Wv2s[(3 * c + d) * EMB + n]);
  } else if (idx < 96 * 512 + 8 * 512) {
    int i2 = idx - 96 * 512;
    int nt = i2 >> 9, within = i2 & 511;
    int lane = within >> 4, t = within & 15;
    int K = 16 * (t >> 3) + 8 * (lane >> 4) + (t & 7);
    int n = nt * 16 + (lane & 15);
    float s = 0.f;
    if (K < 16) {
      for (int k2 = 0; k2 < EMB; ++k2)
        s += Wrbf[K * EMB + k2] * Ws[(2 * EMB + k2) * EMB + n];
    }
    Brbf[i2] = f2bf(s);
  }
}

// -------- main kernel: one wave32 per PAIR of 16-edge tiles, B-pack in LDS --------

__global__ __launch_bounds__(256) void edge_kernel(
    const int* __restrict__ Z, const float* __restrict__ rbf,
    const int* __restrict__ gi, const int* __restrict__ gj,
    const float* __restrict__ R,
    const float* __restrict__ EWi, const float* __restrict__ EWj,
    const float* __restrict__ bfused,
    const float* __restrict__ VWi, const float* __restrict__ VWj,
    const float* __restrict__ bvec,
    const __bf16* __restrict__ Bv2s,   // Brbf contiguous after it
    float* __restrict__ outS, float* __restrict__ outV) {
  extern __shared__ __align__(64) __bf16 lds_b[];   // 104 frags * 512 bf16

  // Cooperative stage of the whole B-pack (106496 B) into LDS: 26 b128 per thread.
  {
    const uint4* src = (const uint4*)Bv2s;
    uint4* dst = (uint4*)lds_b;
#pragma unroll
    for (int it = 0; it < 26; ++it)
      dst[it * 256 + threadIdx.x] = src[it * 256 + threadIdx.x];
  }
  __syncthreads();

  const int lane = threadIdx.x & 31;
  const int wave = threadIdx.x >> 5;
  const int pair = blockIdx.x * 8 + wave;
  if (pair >= N_PAIRS) return;          // uniform per wave (after barrier)

  const int m = lane & 15;              // edge row owned by this lane
  const int h = lane >> 4;              // lane half selects K sub-block

  // Two LDS bases so EVERY fragment access folds to ds_load offset:imm16
  // (no per-fragment VALU address math -> no WMMA WAR-hazard nops).
  const __bf16* lb0 = lds_b + lane * 16;             // frags 0..63
  const __bf16* lb1 = lds_b + 64 * 512 + lane * 16;  // frags 64..103
#define FRAG(f) (*(const v16bf*)(((f) < 64 ? lb0 : lb1) + ((f) & 63) * 512))

  int e[2], zi[2], zj[2];
  float bdv[2][3];
#pragma unroll
  for (int tt = 0; tt < 2; ++tt) {
    e[tt] = (pair * 2 + tt) * 16 + m;
    const int ai = __builtin_nontemporal_load(gi + e[tt]);
    const int aj = __builtin_nontemporal_load(gj + e[tt]);
    zi[tt] = Z[ai]; zj[tt] = Z[aj];
    const float bx = R[aj * 3 + 0] - R[ai * 3 + 0];
    const float by = R[aj * 3 + 1] - R[ai * 3 + 1];
    const float bz = R[aj * 3 + 2] - R[ai * 3 + 2];
    const float d2 = bx * bx + by * by + bz * bz;
#if __has_builtin(__builtin_amdgcn_sqrtf) && __has_builtin(__builtin_amdgcn_rcpf)
    const float inv = __builtin_amdgcn_rcpf(__builtin_amdgcn_sqrtf(d2) + 1e-8f);
#else
    const float inv = 1.0f / (sqrtf(d2) + 1e-8f);
#endif
    bdv[tt][0] = bx * inv; bdv[tt][1] = by * inv; bdv[tt][2] = bz * inv;
  }

  v8f acc[2][8];
  v8f zero8 = {};
#pragma unroll
  for (int tt = 0; tt < 2; ++tt)
#pragma unroll
    for (int nt = 0; nt < 8; ++nt) acc[tt][nt] = zero8;

  // ---- rbf contribution: K=16 padded into one 16x16x32 bf16 WMMA per N-tile
  v16bf bcur = FRAG(96);     // preload first fragment; cvt work below hides latency
  v16bf aR[2];
#pragma unroll
  for (int tt = 0; tt < 2; ++tt) {
    v4f r0 = __builtin_nontemporal_load((const v4f*)(rbf + e[tt] * 16 + 8 * h));
    v4f r1 = __builtin_nontemporal_load((const v4f*)(rbf + e[tt] * 16 + 8 * h) + 1);
    __bf16 zb = f2bf(0.f);
    aR[tt][0] = f2bf(r0[0]); aR[tt][1] = f2bf(r0[1]);
    aR[tt][2] = f2bf(r0[2]); aR[tt][3] = f2bf(r0[3]);
    aR[tt][4] = f2bf(r1[0]); aR[tt][5] = f2bf(r1[1]);
    aR[tt][6] = f2bf(r1[2]); aR[tt][7] = f2bf(r1[3]);
#pragma unroll
    for (int t = 8; t < 16; ++t) aR[tt][t] = zb;   // K=16..31 zero padding
  }
#pragma unroll
  for (int nt = 0; nt < 8; ++nt) {
    v16bf bnxt;
    if (nt < 7) bnxt = FRAG(96 + nt + 1);          // load-ahead
    acc[0][nt] = __builtin_amdgcn_wmma_f32_16x16x32_bf16(
        false, aR[0], false, bcur, (short)0, acc[0][nt], false, false);
    acc[1][nt] = __builtin_amdgcn_wmma_f32_16x16x32_bf16(
        false, aR[1], false, bcur, (short)0, acc[1][nt], false, false);
    if (nt < 7) bcur = bnxt;
  }

  // ---- x_vec build (f32) + three bd-scaled bf16 GEMMs vs W_v2s_d (K=128)
  v4f rsum4[2] = { {}, {} };
#pragma unroll
  for (int kt = 0; kt < 4; ++kt) {
    v4f xv4[2][4];   // 16 x_vec components per tile for this K-tile
#pragma unroll
    for (int tt = 0; tt < 2; ++tt) {
#pragma unroll
      for (int g = 0; g < 2; ++g) {
        const int cb = 32 * kt + 16 * g + 8 * h;
        v4f s0 = *(const v4f*)(bvec + cb);
        v4f s1 = *(const v4f*)(bvec + cb + 4);
#pragma unroll
        for (int d = 0; d < 3; ++d) {
          const float* pi = VWi + (zi[tt] * 3 + d) * EMB + cb;
          const float* pj = VWj + (zj[tt] * 3 + d) * EMB + cb;
          v4f vi0 = *(const v4f*)pi, vi1 = *(const v4f*)(pi + 4);
          v4f vj0 = *(const v4f*)pj, vj1 = *(const v4f*)(pj + 4);
          const float w = bdv[tt][d];
          s0 += w * (vi0 + vj0);
          s1 += w * (vi1 + vj1);
        }
        xv4[tt][2 * g]     = s0;
        xv4[tt][2 * g + 1] = s1;
        rsum4[tt] += s0 + s1;
      }
    }
#pragma unroll
    for (int d = 0; d < 3; ++d) {
      const int fb = (d * 4 + kt) * 8;             // stage's fragment base
      bcur = FRAG(fb);                             // preload while cvt below runs
      v16bf ad[2];
#pragma unroll
      for (int tt = 0; tt < 2; ++tt) {
        const float w = bdv[tt][d];
#pragma unroll
        for (int q = 0; q < 4; ++q) {
          v4f sc = xv4[tt][q] * w;                 // v_pk_mul_f32; fold bd_d into A
          ad[tt][4 * q + 0] = f2bf(sc[0]);
          ad[tt][4 * q + 1] = f2bf(sc[1]);
          ad[tt][4 * q + 2] = f2bf(sc[2]);
          ad[tt][4 * q + 3] = f2bf(sc[3]);
        }
      }
#pragma unroll
      for (int nt = 0; nt < 8; ++nt) {
        v16bf bnxt;
        if (nt < 7) bnxt = FRAG(fb + nt + 1);      // load-ahead
        acc[0][nt] = __builtin_amdgcn_wmma_f32_16x16x32_bf16(
            false, ad[0], false, bcur, (short)0, acc[0][nt], false, false);
        acc[1][nt] = __builtin_amdgcn_wmma_f32_16x16x32_bf16(
            false, ad[1], false, bcur, (short)0, acc[1][nt], false, false);
        if (nt < 7) bcur = bnxt;
      }
    }
  }

  // ---- store: add exact-f32 gathered terms EWi[zi]+EWj[zj]+bfused at write time
#pragma unroll
  for (int tt = 0; tt < 2; ++tt) {
    const int tile = pair * 2 + tt;
    int ziR[8], zjR[8];
#pragma unroll
    for (int r = 0; r < 8; ++r) {
      ziR[r] = __shfl(zi[tt], r + 8 * h, 32);   // C row M = r + 8h lives in lane r+8h
      zjR[r] = __shfl(zj[tt], r + 8 * h, 32);
    }
#pragma unroll
    for (int nt = 0; nt < 8; ++nt) {
      const int n = nt * 16 + m;
      const float bfu = bfused[n];
#pragma unroll
      for (int r = 0; r < 8; ++r) {
        float val = acc[tt][nt][r] + EWi[ziR[r] * EMB + n] + EWj[zjR[r] * EMB + n] + bfu;
        __builtin_nontemporal_store(val, outS + (tile * 16 + 8 * h + r) * EMB + n);
      }
    }
    float rowsum = (rsum4[tt][0] + rsum4[tt][1]) + (rsum4[tt][2] + rsum4[tt][3]);
    const float rtot = rowsum + __shfl_xor(rowsum, 16, 32);
    if (h == 0) {
      __builtin_nontemporal_store(bdv[tt][0] * rtot, outV + e[tt] * 3 + 0);
      __builtin_nontemporal_store(bdv[tt][1] * rtot, outV + e[tt] * 3 + 1);
      __builtin_nontemporal_store(bdv[tt][2] * rtot, outV + e[tt] * 3 + 2);
    }
  }
#undef FRAG
}

// ---------------- launch ----------------

extern "C" void kernel_launch(void* const* d_in, const int* in_sizes, int n_in,
                              void* d_out, int out_size, void* d_ws, size_t ws_size,
                              hipStream_t stream) {
  const int*   Z    = (const int*)  d_in[0];
  const float* rbf  = (const float*)d_in[1];
  const int*   gi   = (const int*)  d_in[2];
  const int*   gj   = (const int*)  d_in[3];
  const float* R    = (const float*)d_in[4];
  const float* emb  = (const float*)d_in[5];
  const float* vemb = (const float*)d_in[6];
  const float* Wrbf = (const float*)d_in[7];
  const float* brbf = (const float*)d_in[8];
  const float* Ws   = (const float*)d_in[9];
  const float* bs   = (const float*)d_in[10];
  const float* Wv   = (const float*)d_in[11];
  const float* bv   = (const float*)d_in[12];
  const float* Wv2s = (const float*)d_in[13];

  char* ws = (char*)d_ws;
  float*  EWi    = (float*)(ws + WS_EWI);
  float*  EWj    = (float*)(ws + WS_EWJ);
  float*  bfused = (float*)(ws + WS_BFUSED);
  float*  VWi    = (float*)(ws + WS_VWI);
  float*  VWj    = (float*)(ws + WS_VWJ);
  __bf16* Bv2s   = (__bf16*)(ws + WS_BV2S);
  __bf16* Brbf   = (__bf16*)(ws + WS_BRBF);

  setup_ew<<<(2 * NZ * EMB + EMB + 255) / 256, 256, 0, stream>>>(
      emb, Ws, brbf, bs, EWi, EWj, bfused);
  setup_vw<<<(2 * NZ * 3 * EMB + 255) / 256, 256, 0, stream>>>(
      vemb, Wv, VWi, VWj);
  setup_bpack<<<(96 * 512 + 8 * 512 + 255) / 256, 256, 0, stream>>>(
      Wv2s, Wrbf, Ws, Bv2s, Brbf);

  float* outS = (float*)d_out;
  float* outV = outS + (size_t)N_EDGES * EMB;
  edge_kernel<<<(N_PAIRS + 7) / 8, 256, LDS_B_BYTES, stream>>>(
      Z, rbf, gi, gj, R, EWi, EWj, bfused, VWi, VWj, bv, Bv2s, outS, outV);
}
